// Pair2mol_50044958933076
// MI455X (gfx1250) — matrix operand order repaired and checked by reading, not compiled, not run on hardware
//
#include <hip/hip_runtime.h>

typedef __attribute__((ext_vector_type(16))) __bf16 v16bf;
typedef __attribute__((ext_vector_type(8)))  float  v8f;
typedef __attribute__((ext_vector_type(4)))  unsigned uv4;
typedef unsigned short u16;

#define DEV __device__ __forceinline__

// ---------- scalar helpers ----------
DEV u16 f2bf(float f) {
  union { float f; unsigned u; } v; v.f = f;
  unsigned r = (v.u + 0x7FFFu + ((v.u >> 16) & 1u)) >> 16;
  return (u16)r;
}
DEV float bf2f(u16 h) {
  union { unsigned u; float f; } v; v.u = ((unsigned)h) << 16;
  return v.f;
}
DEV float sigm(float x) { return 1.0f / (1.0f + __expf(-x)); }

// ---------- WMMA fragment builders (CDNA5 wave32 layouts) ----------
// A lane's 16 bf16 elements are two contiguous 16B runs -> two b128 loads.
union FragBF {
  v16bf v;
  uv4 q[2];
};

// A fragment 16x32 bf16, element (m,k) at base[m*stride + k]; rows 16B-aligned
DEV v16bf load_a_u16(const u16* base, int stride, int lane) {
  FragBF f;
  int row = lane & 15;
  int koff = (lane < 16) ? 0 : 8;
  const u16* p = base + row * stride + koff;
  f.q[0] = *(const uv4*)(p);        // K = koff .. koff+7
  f.q[1] = *(const uv4*)(p + 16);   // K = koff+16 .. koff+23
  return f.v;
}
// B fragment 32x16, element (k,n) at base[n*stride + k] (columns contiguous in k)
DEV v16bf load_b_nmaj(const u16* base, int stride, int lane) {
  FragBF f;
  int col = lane & 15;
  int kb = (lane < 16) ? 0 : 8;
  const u16* p = base + col * stride + kb;
  f.q[0] = *(const uv4*)(p);        // K = kb .. kb+7
  f.q[1] = *(const uv4*)(p + 16);   // K = kb+16 .. kb+23
  return f.v;
}
DEV v8f wmma_bf16(v16bf a, v16bf b, v8f c) {
  return __builtin_amdgcn_wmma_f32_16x16x32_bf16(false, a, false, b, (short)0, c, false, false);
}

// ---------- constants ----------
#define NDIM 256
#define MDIM 512
#define CDIM 128
#define NHEAD 4
#define HSZ 32
#define NNR 65536    // N*N rows
#define NMR 131072   // N*M rows

// ======================================================================
// constant fill (dummy LN vectors: zero mean / unit rstd / gamma=1 / beta=0)
// ======================================================================
__global__ void fill_const(float* __restrict__ p, float v, int n) {
  int i = blockIdx.x * 256 + threadIdx.x;
  if (i < n) p[i] = v;
}

// ======================================================================
// scaling factor: S = sum_m z_mask[0,0,m]; store S^{-1/4}
// ======================================================================
__global__ void scal_kernel(const float* __restrict__ zmask, float* __restrict__ scal) {
  __shared__ float red[512];
  int t = threadIdx.x;
  red[t] = zmask[t];
  __syncthreads();
  for (int s = 256; s > 0; s >>= 1) {
    if (t < s) red[t] += red[t + s];
    __syncthreads();
  }
  if (t == 0) {
    float S = red[0];
    scal[0] = (S > 0.f) ? __powf(S, -0.25f) : 1.0f;
  }
}

// ======================================================================
// per-row LayerNorm stats over 128 channels (one wave per row, wave32)
// ======================================================================
__global__ void row_stats(const float* __restrict__ X, float* __restrict__ mean,
                          float* __restrict__ rstd, int R) {
  int wave = threadIdx.x >> 5, lane = threadIdx.x & 31;
  int row = blockIdx.x * 8 + wave;
  if (row >= R) return;
  const float* p = X + (size_t)row * CDIM;
  float x0 = p[lane], x1 = p[lane + 32], x2 = p[lane + 64], x3 = p[lane + 96];
  float s = x0 + x1 + x2 + x3;
#pragma unroll
  for (int m = 16; m > 0; m >>= 1) s += __shfl_xor(s, m, 32);
  float mu = s * (1.0f / 128.0f);
  float d0 = x0 - mu, d1 = x1 - mu, d2 = x2 - mu, d3 = x3 - mu;
  float v = d0 * d0 + d1 * d1 + d2 * d2 + d3 * d3;
#pragma unroll
  for (int m = 16; m > 0; m >>= 1) v += __shfl_xor(v, m, 32);
  if (lane == 0) {
    mean[row] = mu;
    rstd[row] = rsqrtf(v * (1.0f / 128.0f) + 1e-5f);
  }
}

// ======================================================================
// Generic GEMM: out[R,C] = act( LN(A)[R,K] @ W[K,C] * alpha + bias )
// LN always folded into the A-tile load (dummy vectors => identity).
// 128 threads = 4 waves, 64x64 tile, K-step 32, bf16 WMMA.
// LDS W tile stored TRANSPOSED so B fragments are two ds_load_b128.
// ======================================================================
__global__ void gemm_ln_wmma(const float* __restrict__ A,
                             const float* __restrict__ rmean, const float* __restrict__ rrstd,
                             const float* __restrict__ lng, const float* __restrict__ lnb,
                             const float* __restrict__ W, const float* __restrict__ bias,
                             float* __restrict__ outF, u16* __restrict__ outB,
                             int K, int C, float alpha, int relu) {
  __shared__ u16 As[64][40];   // [row][k]   rows 80B => 16B-aligned
  __shared__ u16 Bs[64][40];   // [col][k]   transposed W tile
  int tid = threadIdx.x;
  int wave = tid >> 5, lane = tid & 31;
  size_t rowBase = (size_t)blockIdx.x * 64;
  int colBase = blockIdx.y * 64;

  v8f acc[4];
#pragma unroll
  for (int f = 0; f < 4; ++f)
#pragma unroll
    for (int v = 0; v < 8; ++v) acc[f][v] = 0.0f;

  for (int kc = 0; kc < K; kc += 32) {
#pragma unroll
    for (int t = tid; t < 2048; t += 128) {
      int r = t >> 5, k = t & 31;
      size_t gr = rowBase + r;
      int gk = kc + k;
      float x = A[gr * K + gk];
      x = (x - rmean[gr]) * rrstd[gr] * lng[gk] + lnb[gk];
      As[r][k] = f2bf(x);
    }
#pragma unroll
    for (int t = tid; t < 2048; t += 128) {
      int k = t >> 6, c = t & 63;
      int gc = colBase + c;
      float wv = (gc < C) ? W[(size_t)(kc + k) * C + gc] : 0.0f;
      Bs[c][k] = f2bf(wv);
    }
    if (kc + 32 < K)  // prefetch next A K-slab (global_prefetch_b8)
      __builtin_prefetch(&A[(rowBase + (tid & 63)) * K + kc + 32], 0, 1);
    __syncthreads();
    v16bf a = load_a_u16(&As[wave * 16][0], 40, lane);
#pragma unroll
    for (int f = 0; f < 4; ++f) {
      v16bf b = load_b_nmaj(&Bs[f * 16][0], 40, lane);
      acc[f] = wmma_bf16(a, b, acc[f]);
    }
    __syncthreads();
  }

  int half = lane >> 4, n = lane & 15;
#pragma unroll
  for (int f = 0; f < 4; ++f) {
    int col = colBase + f * 16 + n;
    if (col >= C) continue;
#pragma unroll
    for (int v = 0; v < 8; ++v) {
      size_t row = rowBase + wave * 16 + v + 8 * half;
      float y = acc[f][v] * alpha + bias[col];
      if (relu) y = fmaxf(y, 0.0f);
      if (outB) outB[row * C + col] = f2bf(y);
      else outF[row * C + col] = y;
    }
  }
}

// ======================================================================
// a = sigmoid(G)*L*zmask*S^{-1/4}; pack bf16 channel-major [c][i][m]
// ======================================================================
__global__ void ew_zpack(const float* __restrict__ G, const float* __restrict__ L,
                         const float* __restrict__ zmask, const float* __restrict__ scal,
                         u16* __restrict__ abf) {
  int r = blockIdx.x, c = threadIdx.x;
  size_t idx = (size_t)r * CDIM + c;
  float v = sigm(G[idx]) * L[idx] * zmask[r] * scal[0];
  int i = r >> 9, m = r & 511;
  abf[((size_t)c * NDIM + i) * MDIM + m] = f2bf(v);
}

// ======================================================================
// z_for_mol[i,j,c] = sum_m a[c][i][m]*a[c][j][m]  (per-channel gram, WMMA)
// ======================================================================
__global__ void gram_wmma(const u16* __restrict__ abf, float* __restrict__ zfm) {
  int c = blockIdx.y;
  int wave = threadIdx.x >> 5, lane = threadIdx.x & 31;
  int t = blockIdx.x * 8 + wave;
  int ti = t >> 4, tj = t & 15;
  const u16* base = abf + (size_t)c * NDIM * MDIM;

  v8f acc;
#pragma unroll
  for (int v = 0; v < 8; ++v) acc[v] = 0.0f;

  for (int mc = 0; mc < MDIM; mc += 32) {
    v16bf a = load_a_u16(base + (size_t)(ti * 16) * MDIM + mc, MDIM, lane);
    v16bf b = load_b_nmaj(base + (size_t)(tj * 16) * MDIM + mc, MDIM, lane);
    acc = wmma_bf16(a, b, acc);
  }
  int half = lane >> 4, n = lane & 15;
#pragma unroll
  for (int v = 0; v < 8; ++v) {
    int i = ti * 16 + v + 8 * half;
    int j = tj * 16 + n;
    zfm[((size_t)i * NDIM + j) * CDIM + c] = acc[v];
  }
}

// ======================================================================
// MP = mol_pair + sigmoid(GM)*U
// ======================================================================
__global__ void ew_update(const float* __restrict__ molp, const float* __restrict__ GM,
                          const float* __restrict__ U, float* __restrict__ MP) {
  size_t idx = (size_t)blockIdx.x * blockDim.x + threadIdx.x;
  MP[idx] = molp[idx] + sigm(GM[idx]) * U[idx];
}

// ======================================================================
// ab = sigmoid(G)*L*mask; pack both [c][i][k] and [c][k][i] bf16 layouts
// ======================================================================
__global__ void ew_abpack(const float* __restrict__ G, const float* __restrict__ L,
                          const float* __restrict__ mask,
                          u16* __restrict__ abn, u16* __restrict__ abt) {
  int r = blockIdx.x, c = threadIdx.x;
  size_t idx = (size_t)r * CDIM + c;
  float v = sigm(G[idx]) * L[idx] * mask[r];
  u16 h = f2bf(v);
  int i = r >> 8, k = r & 255;
  abn[(size_t)c * NNR + r] = h;
  abt[(size_t)c * NNR + (size_t)k * NDIM + i] = h;
}

// ======================================================================
// block[i,j,c] = sum_k ab1[i,k,c]ab2[j,k,c] + sum_k ab1[k,i,c]ab2[k,j,c]
// ======================================================================
__global__ void trigram_wmma(const u16* __restrict__ a1n, const u16* __restrict__ a2n,
                             const u16* __restrict__ a1t, const u16* __restrict__ a2t,
                             float* __restrict__ BLK) {
  int c = blockIdx.y;
  int wave = threadIdx.x >> 5, lane = threadIdx.x & 31;
  int t = blockIdx.x * 8 + wave;
  int ti = t >> 4, tj = t & 15;
  size_t cb = (size_t)c * NNR;

  v8f acc;
#pragma unroll
  for (int v = 0; v < 8; ++v) acc[v] = 0.0f;

  for (int kc = 0; kc < NDIM; kc += 32) {
    v16bf a = load_a_u16(a1n + cb + (size_t)(ti * 16) * NDIM + kc, NDIM, lane);
    v16bf b = load_b_nmaj(a2n + cb + (size_t)(tj * 16) * NDIM + kc, NDIM, lane);
    acc = wmma_bf16(a, b, acc);
  }
  for (int kc = 0; kc < NDIM; kc += 32) {
    v16bf a = load_a_u16(a1t + cb + (size_t)(ti * 16) * NDIM + kc, NDIM, lane);
    v16bf b = load_b_nmaj(a2t + cb + (size_t)(tj * 16) * NDIM + kc, NDIM, lane);
    acc = wmma_bf16(a, b, acc);
  }
  int half = lane >> 4, n = lane & 15;
#pragma unroll
  for (int v = 0; v < 8; ++v) {
    int i = ti * 16 + v + 8 * half;
    int j = tj * 16 + n;
    BLK[((size_t)i * NDIM + j) * CDIM + c] = acc[v];
  }
}

// ======================================================================
// MP += sigmoid(GE)*AF*mask
// ======================================================================
__global__ void ew_gatemaskadd(float* __restrict__ MP, const float* __restrict__ GE,
                               const float* __restrict__ AF, const float* __restrict__ mask) {
  size_t idx = (size_t)blockIdx.x * blockDim.x + threadIdx.x;
  MP[idx] += sigm(GE[idx]) * AF[idx] * mask[idx >> 7];
}

// ======================================================================
// Triangle attention, one workgroup per (pivot i, head h).
// bf16 logits [256][264] + transposed summed-V [32][264] in LDS (~149KB)
// => 2 workgroups/WGP on 320KB LDS; all fragments are b128 loads.
// ======================================================================
#define LSTR 264
__global__ void attn_wmma(const u16* __restrict__ Q, const u16* __restrict__ Kb,
                          const u16* __restrict__ Vb, const float* __restrict__ BiasF,
                          const float* __restrict__ mask, float* __restrict__ Out) {
  extern __shared__ char smem[];
  u16* Lg = (u16*)smem;                                   // [256][LSTR] bf16 logits/probs
  u16* Vt = (u16*)(smem + (size_t)NDIM * LSTR * sizeof(u16)); // [32][LSTR] Vt[c][k]

  int i = blockIdx.x, h = blockIdx.y;
  int tid = threadIdx.x;
  int wave = tid >> 5, lane = tid & 31;
  int half = lane >> 4, nn = lane & 15;

  // ---- phase 1: logits W1 + W2 + bias + bias^T + mask ----
  for (int tt = 0; tt < 32; ++tt) {
    int t = tt * 8 + wave;
    int tq = t >> 4, tk = t & 15;
    v8f acc;
#pragma unroll
    for (int v = 0; v < 8; ++v) acc[v] = 0.0f;
    // W1: rows (i,q), cols (i,k), contract c (exactly one WMMA, K=32)
    v16bf a1 = load_a_u16(Q + ((size_t)(i * NDIM + tq * 16)) * CDIM + h * HSZ, CDIM, lane);
    v16bf b1 = load_b_nmaj(Kb + ((size_t)(i * NDIM + tk * 16)) * CDIM + h * HSZ, CDIM, lane);
    acc = wmma_bf16(a1, b1, acc);
    // W2: rows (q,i), cols (k,i)
    v16bf a2 = load_a_u16(Q + ((size_t)(tq * 16) * NDIM + i) * CDIM + h * HSZ, NDIM * CDIM, lane);
    v16bf b2 = load_b_nmaj(Kb + ((size_t)(tk * 16) * NDIM + i) * CDIM + h * HSZ, NDIM * CDIM, lane);
    acc = wmma_bf16(a2, b2, acc);
#pragma unroll
    for (int v = 0; v < 8; ++v) {
      int q = tq * 16 + v + 8 * half;
      int kx = tk * 16 + nn;
      float val = acc[v]
                + BiasF[((size_t)q * NDIM + kx) * NHEAD + h]
                + BiasF[((size_t)kx * NDIM + q) * NHEAD + h]
                + 1e9f * (mask[(size_t)i * NDIM + kx] - 1.0f);
      Lg[q * LSTR + kx] = f2bf(val);
    }
  }
  // transposed summed V: Vt[c][k] = V[i,k,h,c] + V[k,i,h,c]
  for (int idx = tid; idx < HSZ * NDIM; idx += 256) {
    int c = idx >> 8, k = idx & 255;
    float f = bf2f(Vb[((size_t)(i * NDIM + k)) * CDIM + h * HSZ + c])
            + bf2f(Vb[((size_t)k * NDIM + i) * CDIM + h * HSZ + c]);
    Vt[c * LSTR + k] = f2bf(f);
  }
  __syncthreads();

  // ---- phase 2: row softmax over keys (32 rows per wave) ----
  for (int rr = 0; rr < 32; ++rr) {
    int q = rr * 8 + wave;
    u16* row = Lg + q * LSTR;
    float x[8];
    float mx = -3.4e38f;
#pragma unroll
    for (int j = 0; j < 8; ++j) {
      x[j] = bf2f(row[lane + 32 * j]);
      mx = fmaxf(mx, x[j]);
    }
#pragma unroll
    for (int m = 16; m > 0; m >>= 1) mx = fmaxf(mx, __shfl_xor(mx, m, 32));
    float s = 0.0f;
#pragma unroll
    for (int j = 0; j < 8; ++j) {
      x[j] = __expf(x[j] - mx);
      s += x[j];
    }
#pragma unroll
    for (int m = 16; m > 0; m >>= 1) s += __shfl_xor(s, m, 32);
    float inv = 1.0f / s;
#pragma unroll
    for (int j = 0; j < 8; ++j) row[lane + 32 * j] = f2bf(x[j] * inv);
  }
  __syncthreads();

  // ---- phase 3: out[q,c] = sum_k W[q,k]*Vs[k,c] ----
  for (int tt = 0; tt < 4; ++tt) {
    int t = tt * 8 + wave;
    int tq = t >> 1, tc = t & 1;
    v8f acc;
#pragma unroll
    for (int v = 0; v < 8; ++v) acc[v] = 0.0f;
    for (int kc = 0; kc < NDIM; kc += 32) {
      v16bf a = load_a_u16(Lg + (size_t)(tq * 16) * LSTR + kc, LSTR, lane);
      v16bf b = load_b_nmaj(Vt + (size_t)(tc * 16) * LSTR + kc, LSTR, lane);
      acc = wmma_bf16(a, b, acc);
    }
#pragma unroll
    for (int v = 0; v < 8; ++v) {
      int q = tq * 16 + v + 8 * half;
      int c = tc * 16 + nn;
      Out[((size_t)(i * NDIM + q)) * CDIM + h * HSZ + c] = acc[v];
    }
  }
}

// ======================================================================
// Y = sigmoid(G)*Y ;  MP += FA*mask
// ======================================================================
__global__ void ew_sigmul(float* __restrict__ Y, const float* __restrict__ G) {
  size_t idx = (size_t)blockIdx.x * blockDim.x + threadIdx.x;
  Y[idx] = sigm(G[idx]) * Y[idx];
}
__global__ void ew_maskadd(float* __restrict__ MP, const float* __restrict__ FA,
                           const float* __restrict__ mask) {
  size_t idx = (size_t)blockIdx.x * blockDim.x + threadIdx.x;
  MP[idx] += FA[idx] * mask[idx >> 7];
}

// ======================================================================
// host-side orchestration
// ======================================================================
extern "C" void kernel_launch(void* const* d_in, const int* in_sizes, int n_in,
                              void* d_out, int out_size, void* d_ws, size_t ws_size,
                              hipStream_t stream) {
  (void)in_sizes; (void)n_in; (void)out_size; (void)ws_size;
  const float* mol_pair = (const float*)d_in[0];
  const float* mol_mask = (const float*)d_in[1];
  const float* zin      = (const float*)d_in[2];
  const float* z_mask   = (const float*)d_in[3];
  auto P = [&](int i) { return (const float*)d_in[i]; };
  // params in setup_inputs() dict insertion order
  const float *norm_z_g = P(4),  *norm_z_b = P(5);
  const float *gl1_w = P(6),  *gl1_b = P(7),  *l1_w = P(8),  *l1_b = P(9);
  const float *un_g = P(10),  *un_b = P(11),  *ul_w = P(12), *ul_b = P(13);
  const float *gmp_w = P(14), *gmp_b = P(15);
  const float *tu_ln_g = P(16), *tu_ln_b = P(17), *tu_lnh_g = P(18), *tu_lnh_b = P(19);
  const float *tu_gl1_w = P(20), *tu_gl1_b = P(21), *tu_gl2_w = P(22), *tu_gl2_b = P(23);
  const float *tu_l1_w = P(24), *tu_l1_b = P(25), *tu_l2_w = P(26), *tu_l2_b = P(27);
  const float *tu_end_w = P(28), *tu_end_b = P(29), *tu_after_w = P(30), *tu_after_b = P(31);
  const float *ta_ln_g = P(32), *ta_ln_b = P(33);
  const float *ta_q_w = P(34), *ta_k_w = P(35), *ta_v_w = P(36), *ta_bias_w = P(37);
  const float *ta_g_w = P(38), *ta_g_b = P(39), *ta_final_w = P(40), *ta_final_b = P(41);
  const float *tr_ln_g = P(42), *tr_ln_b = P(43);
  const float *tr_l1_w = P(44), *tr_l1_b = P(45), *tr_l2_w = P(46), *tr_l2_b = P(47);

  // workspace arena (reused slots)
  char* ws = (char*)d_ws;
  const size_t SZ_NMf = (size_t)NMR * CDIM * 4;   // 64MB
  const size_t SZ_NNf = (size_t)NNR * CDIM * 4;   // 32MB
  const size_t SZ_NNb = (size_t)NNR * CDIM * 2;   // 16MB
  size_t o = 0;
  char* slot0 = ws + o; o += SZ_NMf;   // G  / T1
  char* slot1 = ws + o; o += SZ_NMf;   // L  / Qbf+Kbf+Vbf+BiasF
  char* slot2 = ws + o; o += SZ_NNf;   // abf / G1,GE / Gf
  char* slot3 = ws + o; o += SZ_NNf;   // zfm / BLK / OutAttn
  char* slot4 = ws + o; o += SZ_NNf;   // U / L1,AF / FA
  char* slot5 = ws + o; o += SZ_NNf;   // GM / ab1n+ab1t
  char* slot6 = ws + o; o += SZ_NNf;   // MP (persistent)
  char* slot7 = ws + o; o += SZ_NNf;   // ab2n+ab2t
  float* meanNM = (float*)(ws + o); o += NMR * 4;
  float* rstdNM = (float*)(ws + o); o += NMR * 4;
  float* meanA  = (float*)(ws + o); o += NNR * 4;
  float* rstdA  = (float*)(ws + o); o += NNR * 4;
  float* meanB  = (float*)(ws + o); o += NNR * 4;
  float* rstdB  = (float*)(ws + o); o += NNR * 4;
  float* scal   = (float*)(ws + o); o += 256;
  float* rowZ   = (float*)(ws + o); o += NMR * 4;   // zeros  (dummy mean)
  float* rowO   = (float*)(ws + o); o += NMR * 4;   // ones   (dummy rstd)
  float* colO   = (float*)(ws + o); o += 256 * 4;   // ones   (dummy gamma)
  float* colZ   = (float*)(ws + o); o += 256 * 4;   // zeros  (dummy beta/bias)

  auto gemm = [&](const float* A, const float* rm, const float* rs,
                  const float* g, const float* b, const float* W, const float* bias,
                  float* oF, u16* oB, int R, int K, int C, float alpha, int relu) {
    dim3 gr(R / 64, (C + 63) / 64);
    gemm_ln_wmma<<<gr, 128, 0, stream>>>(A, rm, rs, g, b, W, bias, oF, oB, K, C, alpha, relu);
  };

  // dummy LN / bias vectors (branch-free GEMM inner loop)
  fill_const<<<NMR / 256, 256, 0, stream>>>(rowZ, 0.0f, NMR);
  fill_const<<<NMR / 256, 256, 0, stream>>>(rowO, 1.0f, NMR);
  fill_const<<<1, 256, 0, stream>>>(colO, 1.0f, 256);
  fill_const<<<1, 256, 0, stream>>>(colZ, 0.0f, 256);

  // ---- z branch: LN + gated linear + gram contraction ----
  scal_kernel<<<1, 512, 0, stream>>>(z_mask, scal);
  row_stats<<<NMR / 8, 256, 0, stream>>>(zin, meanNM, rstdNM, NMR);
  float* G = (float*)slot0;  float* L = (float*)slot1;
  gemm(zin, meanNM, rstdNM, norm_z_g, norm_z_b, gl1_w, gl1_b, G, nullptr, NMR, CDIM, CDIM, 1.f, 0);
  gemm(zin, meanNM, rstdNM, norm_z_g, norm_z_b, l1_w,  l1_b,  L, nullptr, NMR, CDIM, CDIM, 1.f, 0);
  u16* abf = (u16*)slot2;
  ew_zpack<<<NMR, CDIM, 0, stream>>>(G, L, z_mask, scal, abf);
  float* zfm = (float*)slot3;
  gram_wmma<<<dim3(32, CDIM), 256, 0, stream>>>(abf, zfm);

  // ---- mol_pair update: gate * lin(LN(zfm)) ----
  row_stats<<<NNR / 8, 256, 0, stream>>>(zfm, meanB, rstdB, NNR);
  float* U = (float*)slot4;  float* GM = (float*)slot5;  float* MP = (float*)slot6;
  gemm(zfm, meanB, rstdB, un_g, un_b, ul_w, ul_b, U, nullptr, NNR, CDIM, CDIM, 1.f, 0);
  gemm(mol_pair, rowZ, rowO, colO, colZ, gmp_w, gmp_b, GM, nullptr, NNR, CDIM, CDIM, 1.f, 0);
  ew_update<<<(NNR * CDIM) / 256, 256, 0, stream>>>(mol_pair, GM, U, MP);

  // ---- triangle multiplicative update ----
  row_stats<<<NNR / 8, 256, 0, stream>>>(MP, meanA, rstdA, NNR);
  float* G1 = (float*)slot2;  float* L1 = (float*)slot4;
  u16* ab1n = (u16*)slot5;  u16* ab1t = (u16*)(slot5 + SZ_NNb);
  u16* ab2n = (u16*)slot7;  u16* ab2t = (u16*)(slot7 + SZ_NNb);
  gemm(MP, meanA, rstdA, tu_ln_g, tu_ln_b, tu_gl1_w, tu_gl1_b, G1, nullptr, NNR, CDIM, CDIM, 1.f, 0);
  gemm(MP, meanA, rstdA, tu_ln_g, tu_ln_b, tu_l1_w,  tu_l1_b,  L1, nullptr, NNR, CDIM, CDIM, 1.f, 0);
  ew_abpack<<<NNR, CDIM, 0, stream>>>(G1, L1, mol_mask, ab1n, ab1t);
  gemm(MP, meanA, rstdA, tu_ln_g, tu_ln_b, tu_gl2_w, tu_gl2_b, G1, nullptr, NNR, CDIM, CDIM, 1.f, 0);
  gemm(MP, meanA, rstdA, tu_ln_g, tu_ln_b, tu_l2_w,  tu_l2_b,  L1, nullptr, NNR, CDIM, CDIM, 1.f, 0);
  ew_abpack<<<NNR, CDIM, 0, stream>>>(G1, L1, mol_mask, ab2n, ab2t);
  float* BLK = (float*)slot3;
  trigram_wmma<<<dim3(32, CDIM), 256, 0, stream>>>(ab1n, ab2n, ab1t, ab2t, BLK);
  row_stats<<<NNR / 8, 256, 0, stream>>>(BLK, meanB, rstdB, NNR);
  float* GE = (float*)slot2;  float* AF = (float*)slot4;
  gemm(MP,  meanA, rstdA, tu_ln_g,  tu_ln_b,  tu_end_w,   tu_end_b,   GE, nullptr, NNR, CDIM, CDIM, 1.f, 0);
  gemm(BLK, meanB, rstdB, tu_lnh_g, tu_lnh_b, tu_after_w, tu_after_b, AF, nullptr, NNR, CDIM, CDIM, 1.f, 0);
  ew_gatemaskadd<<<(NNR * CDIM) / 256, 256, 0, stream>>>(MP, GE, AF, mol_mask);

  // ---- triangle attention ----
  row_stats<<<NNR / 8, 256, 0, stream>>>(MP, meanA, rstdA, NNR);
  u16* Qbf = (u16*)slot1;
  u16* Kbf = (u16*)(slot1 + SZ_NNb);
  u16* Vbf = (u16*)(slot1 + 2 * SZ_NNb);
  float* BiasF = (float*)(slot1 + 3 * SZ_NNb);
  float* Gf = (float*)slot2;
  const float qscale = 0.17677669529663687f;  // HSZ^-0.5
  gemm(MP, meanA, rstdA, ta_ln_g, ta_ln_b, ta_q_w, colZ, nullptr, Qbf, NNR, CDIM, CDIM, qscale, 0);
  gemm(MP, meanA, rstdA, ta_ln_g, ta_ln_b, ta_k_w, colZ, nullptr, Kbf, NNR, CDIM, CDIM, 1.f, 0);
  gemm(MP, meanA, rstdA, ta_ln_g, ta_ln_b, ta_v_w, colZ, nullptr, Vbf, NNR, CDIM, CDIM, 1.f, 0);
  gemm(MP, meanA, rstdA, ta_ln_g, ta_ln_b, ta_bias_w, colZ, BiasF, nullptr, NNR, CDIM, NHEAD, 1.f, 0);
  gemm(MP, meanA, rstdA, ta_ln_g, ta_ln_b, ta_g_w, ta_g_b, Gf, nullptr, NNR, CDIM, CDIM, 1.f, 0);
  float* OutAttn = (float*)slot3;
  size_t attn_smem = (size_t)(NDIM + HSZ) * LSTR * sizeof(u16);
  attn_wmma<<<dim3(NDIM, NHEAD), 256, attn_smem, stream>>>(Qbf, Kbf, Vbf, BiasF, mol_mask, OutAttn);
  ew_sigmul<<<(NNR * CDIM) / 256, 256, 0, stream>>>(OutAttn, Gf);
  float* FA = (float*)slot4;
  gemm(OutAttn, rowZ, rowO, colO, colZ, ta_final_w, ta_final_b, FA, nullptr, NNR, CDIM, CDIM, 1.f, 0);
  ew_maskadd<<<(NNR * CDIM) / 256, 256, 0, stream>>>(MP, FA, mol_mask);

  // ---- transition MLP -> d_out ----
  row_stats<<<NNR / 8, 256, 0, stream>>>(MP, meanB, rstdB, NNR);
  float* T1 = (float*)slot0;
  gemm(MP, meanB, rstdB, tr_ln_g, tr_ln_b, tr_l1_w, tr_l1_b, T1, nullptr, NNR, CDIM, 2 * CDIM, 1.f, 1);
  gemm(T1, rowZ, rowO, colO, colZ, tr_l2_w, tr_l2_b, (float*)d_out, nullptr, NNR, 2 * CDIM, CDIM, 1.f, 0);
}
